// smallGraphAlignLayer_42528766165730
// MI455X (gfx1250) — matrix operand order repaired and checked by reading, not compiled, not run on hardware
//
#include <hip/hip_runtime.h>
#include <cstdint>

#define NN  20000
#define KN  8
#define INF 128
#define HH  128
#define WV2 2

typedef __attribute__((ext_vector_type(16))) __bf16 v16bf;
typedef __attribute__((ext_vector_type(8)))  float  v8f;
typedef unsigned int u32x4 __attribute__((__vector_size__(16)));
typedef int          i32x8 __attribute__((__vector_size__(32)));
typedef int          i32x4 __attribute__((__vector_size__(16)));

#if defined(__HIP_DEVICE_COMPILE__) && __has_builtin(__builtin_amdgcn_global_load_async_to_lds_b128) && __has_builtin(__builtin_amdgcn_s_wait_asynccnt)
#define USE_ASYNC_LDS 1
#else
#define USE_ASYNC_LDS 0
#endif

#if defined(__HIP_DEVICE_COMPILE__) && __has_builtin(__builtin_amdgcn_tensor_load_to_lds) && __has_builtin(__builtin_amdgcn_s_wait_tensorcnt)
#define USE_TDM 1
#else
#define USE_TDM 0
#endif

#if USE_ASYNC_LDS
typedef int v4i __attribute__((__vector_size__(16)));
typedef __attribute__((address_space(1))) v4i gv4i;   // global int4
typedef __attribute__((address_space(3))) v4i lv4i;   // LDS int4
// Generic LDS pointers carry the DS byte offset in their low 32 bits (ISA 10.2);
// global generic pointers are value-identical to as(1).
__device__ __forceinline__ gv4i* as_global_v4(const void* p) { return (gv4i*)(uintptr_t)p; }
__device__ __forceinline__ lv4i* as_lds_v4(const void* p)    { return (lv4i*)(uint32_t)(uintptr_t)p; }
#endif

__device__ __forceinline__ float relu_f(float x) { return x > 0.f ? x : 0.f; }

// ---------------------------------------------------------------------------
// Kernel 1: self_h = feat@W0+b0, Wh1 = feat@W1+b1, Wh2 = feat@W2+b2
// 8 waves/block, each wave owns a 16-row strip. Weight panel staged via TDM
// (tensor_load_to_lds, 64x128 f32 halves) then transposed to bf16 in LDS.
// ---------------------------------------------------------------------------
__global__ void __launch_bounds__(256)
gemm3_wmma(const float* __restrict__ feat,
           const float* __restrict__ W0, const float* __restrict__ B0,
           const float* __restrict__ W1, const float* __restrict__ B1,
           const float* __restrict__ W2, const float* __restrict__ B2,
           float* __restrict__ selfh, float* __restrict__ Wh1, float* __restrict__ Wh2)
{
  __shared__ __attribute__((aligned(32))) __bf16 sW[INF * HH];   // transposed: sW[n*128 + k]
#if USE_TDM
  __shared__ __attribute__((aligned(16))) float sWf[64 * INF];   // 32KB f32 staging (half panel)
#endif

  const int tid  = threadIdx.x;
  const int wv   = tid >> 5;
  const int lane = tid & 31;
  const int r    = lane & 15;
  const int hs   = lane >> 4;
  const int m0   = blockIdx.x * 128 + wv * 16;
  int arow = m0 + r;
  if (arow >= NN) arow = NN - 1;

  // A fragments: 16x128 row strip of feat as bf16, element s holds k = ks*32 + 16*hs + s (same order as B)
  v16bf af[4];
  #pragma unroll
  for (int ks = 0; ks < 4; ++ks) {
    const float* p = feat + (size_t)arow * INF + ks * 32 + hs * 16;
    #pragma unroll
    for (int j = 0; j < 8; ++j) {
      af[ks][j]     = (__bf16)p[j];
      af[ks][8 + j] = (__bf16)p[8 + j];
    }
  }

  const float* Wm[3] = {W0, W1, W2};
  const float* Bm[3] = {B0, B1, B2};
  float*       Dm[3] = {selfh, Wh1, Wh2};

  #pragma unroll
  for (int mat = 0; mat < 3; ++mat) {
    const float* Wg = Wm[mat];

#if USE_TDM
    #pragma unroll
    for (int h = 0; h < 2; ++h) {
      __syncthreads();                      // sWf free / prev-mat sW reads done
      if (wv == 0) {
        const uint64_t ga = (uint64_t)(uintptr_t)(Wg + (size_t)h * 64 * INF);
        const uint32_t la = (uint32_t)(uintptr_t)(&sWf[0]);
        // D# group0: count=1 | lds_addr | global_addr[56:0] | type=2  (ISA 8.3)
        u32x4 g0 = { 1u, la, (uint32_t)ga,
                     (uint32_t)((ga >> 32) & 0x01FFFFFFu) | 0x80000000u };
        // D# group1: data_size=4B; tensor_dim0=128, tensor_dim1=64; tile 128x64;
        // dim0_stride=128, dim1_stride=8192  (ISA 8.4 bit packing)
        i32x8 g1 = { (int)0x00020000,
                     (int)(128u << 16),     // tensor_dim0[15:0] << 16
                     (int)(64u  << 16),     // tensor_dim1[15:0] << 16
                     (int)(128u << 16),     // tile_dim0 << 16
                     (int)64,               // tile_dim1
                     (int)128,              // tensor_dim0_stride[31:0]
                     (int)(8192u << 16),    // tensor_dim1_stride[15:0] << 16
                     0 };
        i32x4 gz4 = {0, 0, 0, 0};
        i32x8 gz8 = {0, 0, 0, 0, 0, 0, 0, 0};
        __builtin_amdgcn_tensor_load_to_lds(g0, g1, gz4, gz4, gz8, 0);
        __builtin_amdgcn_s_wait_tensorcnt(0);
      }
      __syncthreads();
      // convert staged f32 rows k = h*64 .. h*64+63 into transposed bf16 panel
      for (int idx = tid; idx < 64 * INF; idx += 256) {
        const int kk = idx >> 7;
        const int n  = idx & 127;
        sW[n * INF + h * 64 + kk] = (__bf16)sWf[idx];
      }
    }
    __syncthreads();
#else
    __syncthreads();
    for (int idx = tid; idx < INF * HH; idx += 256) {
      const int k = idx >> 7;
      const int n = idx & 127;
      sW[n * INF + k] = (__bf16)Wg[idx];
    }
    __syncthreads();
#endif

    if (mat < 2) __builtin_prefetch(Wm[mat + 1] + tid * 64, 0, 0);

    #pragma unroll
    for (int nt = 0; nt < 8; ++nt) {
      v8f acc = {0.f, 0.f, 0.f, 0.f, 0.f, 0.f, 0.f, 0.f};
      #pragma unroll
      for (int ks = 0; ks < 4; ++ks) {
        const __bf16* bp = &sW[(nt * 16 + r) * INF + ks * 32 + hs * 16];
        v16bf bf = *(const v16bf*)bp;
        acc = __builtin_amdgcn_wmma_f32_16x16x32_bf16(false, af[ks], false, bf,
                                                      (short)0, acc, false, false);
      }
      const int   col  = nt * 16 + r;
      const float bias = Bm[mat][col];
      float*      D    = Dm[mat];
      #pragma unroll
      for (int v = 0; v < 8; ++v) {
        const int rr = m0 + v + 8 * hs;
        if (rr < NN) D[(size_t)rr * HH + col] = acc[v] + bias;
      }
    }
  }
}

// ---------------------------------------------------------------------------
// Kernel 2: per-node attention aggregation. One wave per node.
// Gather of 32 rows done with async global->LDS DMA; gram G = Z*M^T via WMMA.
// ---------------------------------------------------------------------------
__global__ void __launch_bounds__(WV2 * 32)
gat_attn_wmma(const float* __restrict__ Wh1, const float* __restrict__ Wh2,
              const float* __restrict__ selfh, const float* __restrict__ attn,
              const int* __restrict__ src1, const int* __restrict__ src2,
              float* __restrict__ out)
{
  __shared__ float sAttn[2 * HH];
  __shared__ __attribute__((aligned(16))) float sZ[WV2][16 * HH];
  __shared__ __attribute__((aligned(16))) float sM[WV2][16 * HH];
  __shared__ float sRed[WV2][96]; // znorm | mnorm | msum | e_d | bw_raw | w_final

  const int tid  = threadIdx.x;
  const int wv   = tid >> 5;
  const int lane = tid & 31;
  int node = blockIdx.x * WV2 + wv;
  if (node >= NN) node = NN - 1;

  for (int c = tid; c < 2 * HH; c += WV2 * 32) sAttn[c] = attn[c];

  const int* s1 = src1 + (size_t)node * KN;
  const int* s2 = src2 + (size_t)node * KN;
  const int  c0 = lane * 4;

  // Gather 32 rows: Z = [Wh1[src1] ; Wh2[src2]], M = [Wh1[src2] ; Wh2[src1]]
#if USE_ASYNC_LDS
  #pragma unroll
  for (int rrow = 0; rrow < 8; ++rrow) {
    const int i1 = s1[rrow];
    const int i2 = s2[rrow];
    __builtin_amdgcn_global_load_async_to_lds_b128(
        as_global_v4(Wh1 + (size_t)i1 * HH + c0), as_lds_v4(&sZ[wv][rrow * HH + c0]), 0, 0);
    __builtin_amdgcn_global_load_async_to_lds_b128(
        as_global_v4(Wh2 + (size_t)i2 * HH + c0), as_lds_v4(&sZ[wv][(rrow + 8) * HH + c0]), 0, 0);
    __builtin_amdgcn_global_load_async_to_lds_b128(
        as_global_v4(Wh1 + (size_t)i2 * HH + c0), as_lds_v4(&sM[wv][rrow * HH + c0]), 0, 0);
    __builtin_amdgcn_global_load_async_to_lds_b128(
        as_global_v4(Wh2 + (size_t)i1 * HH + c0), as_lds_v4(&sM[wv][(rrow + 8) * HH + c0]), 0, 0);
  }
  __builtin_amdgcn_s_wait_asynccnt(0);
#else
  #pragma unroll
  for (int rrow = 0; rrow < 8; ++rrow) {
    const int i1 = s1[rrow];
    const int i2 = s2[rrow];
    float4 a = *(const float4*)(Wh1 + (size_t)i1 * HH + c0);
    float4 b = *(const float4*)(Wh2 + (size_t)i2 * HH + c0);
    float4 c = *(const float4*)(Wh1 + (size_t)i2 * HH + c0);
    float4 d = *(const float4*)(Wh2 + (size_t)i1 * HH + c0);
    *(float4*)&sZ[wv][rrow * HH + c0]       = a;
    *(float4*)&sZ[wv][(rrow + 8) * HH + c0] = b;
    *(float4*)&sM[wv][rrow * HH + c0]       = c;
    *(float4*)&sM[wv][(rrow + 8) * HH + c0] = d;
  }
#endif

  float4 sh = *(const float4*)(selfh + (size_t)node * HH + c0);

  __syncthreads();

  // eh = self_h[n] . a_h
  float ehp = sh.x * sAttn[HH + c0] + sh.y * sAttn[HH + c0 + 1] +
              sh.z * sAttn[HH + c0 + 2] + sh.w * sAttn[HH + c0 + 3];
  #pragma unroll
  for (int off = 16; off > 0; off >>= 1) ehp += __shfl_xor(ehp, off, 32);
  const float eh = ehp;

  // Half-wave specialization: lanes 0-15 -> Z norms + z.a_z; lanes 16-31 -> M norms + sums
  const int rsel = lane & 15;
  const float* rowp = (lane < 16) ? &sZ[wv][rsel * HH] : &sM[wv][rsel * HH];
  float nrm = 0.f, dot = 0.f, ssum = 0.f;
  for (int c = 0; c < HH; c += 4) {
    float4 v  = *(const float4*)(rowp + c);
    float4 az = *(const float4*)(&sAttn[c]);
    nrm  += v.x * v.x + v.y * v.y + v.z * v.z + v.w * v.w;
    dot  += v.x * az.x + v.y * az.y + v.z * az.z + v.w * az.w;
    ssum += v.x + v.y + v.z + v.w;
  }
  if (lane < 16) {
    sRed[wv][rsel] = nrm;
    float e = dot + eh;
    sRed[wv][48 + rsel] = (e > 0.f) ? e : 0.01f * e;   // leaky_relu 0.01
  } else {
    sRed[wv][16 + rsel] = nrm;
    sRed[wv][32 + rsel] = ssum;
  }

  __syncthreads();

  // Gram matrix G = Z * M^T via WMMA (bf16 in, f32 acc)
  const int hs = lane >> 4;
  v8f acc = {0.f, 0.f, 0.f, 0.f, 0.f, 0.f, 0.f, 0.f};
  #pragma unroll
  for (int ks = 0; ks < 4; ++ks) {
    const int k0 = ks * 32;
    v16bf afr, bfr;
    const float* zp = &sZ[wv][rsel * HH + k0 + hs * 16];
    #pragma unroll
    for (int j = 0; j < 8; ++j) { afr[j] = (__bf16)zp[j]; afr[8 + j] = (__bf16)zp[8 + j]; }
    const float* mp = &sM[wv][rsel * HH + k0 + hs * 16];
    #pragma unroll
    for (int j = 0; j < 16; ++j) bfr[j] = (__bf16)mp[j];
    acc = __builtin_amdgcn_wmma_f32_16x16x32_bf16(false, afr, false, bfr,
                                                  (short)0, acc, false, false);
  }

  // aw = gate_e * exp(-sqrt(max(|z_d|^2 + |m_e|^2 - 2 G, 1e-12))); bw_raw_d = sum_e aw
  const float mn   = sRed[wv][16 + rsel];
  const float ms   = sRed[wv][32 + rsel];
  const float gate = (ms != 0.f) ? 1.f : 0.f;
  float awv[8];
  #pragma unroll
  for (int v = 0; v < 8; ++v) {
    const int d = v + 8 * hs;
    float d2 = sRed[wv][d] + mn - 2.f * acc[v];
    d2 = fmaxf(d2, 1e-12f);
    awv[v] = gate * __expf(-sqrtf(d2));
  }
  #pragma unroll
  for (int off = 1; off < 16; off <<= 1) {
    #pragma unroll
    for (int v = 0; v < 8; ++v) awv[v] += __shfl_xor(awv[v], off, 32);
  }
  if (rsel == 0) {
    #pragma unroll
    for (int v = 0; v < 8; ++v) sRed[wv][64 + v + 8 * hs] = awv[v];
  }

  __syncthreads();

  // Final weights: w_d ∝ bw_raw_d * exp(e_d); all normalization factors cancel
  {
    const float e   = sRed[wv][48 + rsel];
    const float bwr = sRed[wv][64 + rsel];
    float mx = e;
    #pragma unroll
    for (int off = 1; off < 16; off <<= 1) mx = fmaxf(mx, __shfl_xor(mx, off, 32));
    float wd = bwr * __expf(e - mx);
    float s = wd;
    #pragma unroll
    for (int off = 1; off < 16; off <<= 1) s += __shfl_xor(s, off, 32);
    wd = wd / s;
    if (lane < 16) sRed[wv][80 + lane] = wd;
  }

  __syncthreads();

  // agg = sum_d w_d * Z_d ; output relu(concat(self_h, agg))
  float4 agg = {0.f, 0.f, 0.f, 0.f};
  #pragma unroll
  for (int d = 0; d < 16; ++d) {
    const float wd = sRed[wv][80 + d];
    float4 z = *(const float4*)(&sZ[wv][d * HH + c0]);
    agg.x += wd * z.x; agg.y += wd * z.y; agg.z += wd * z.z; agg.w += wd * z.w;
  }

  float* op = out + (size_t)node * (2 * HH);
  float4 o1 = { relu_f(sh.x),  relu_f(sh.y),  relu_f(sh.z),  relu_f(sh.w)  };
  float4 o2 = { relu_f(agg.x), relu_f(agg.y), relu_f(agg.z), relu_f(agg.w) };
  *(float4*)(op + c0)      = o1;
  *(float4*)(op + HH + c0) = o2;
}

// ---------------------------------------------------------------------------
extern "C" void kernel_launch(void* const* d_in, const int* in_sizes, int n_in,
                              void* d_out, int out_size, void* d_ws, size_t ws_size,
                              hipStream_t stream) {
  const float* feat = (const float*)d_in[0];
  const float* W0   = (const float*)d_in[1];
  const float* B0   = (const float*)d_in[2];
  const float* W1   = (const float*)d_in[3];
  const float* B1   = (const float*)d_in[4];
  const float* W2   = (const float*)d_in[5];
  const float* B2   = (const float*)d_in[6];
  const float* attn = (const float*)d_in[7];
  const int*   src1 = (const int*)d_in[8];
  const int*   src2 = (const int*)d_in[9];
  float*       outp = (float*)d_out;

  float* selfh = (float*)d_ws;
  float* Wh1   = selfh + (size_t)NN * HH;
  float* Wh2   = Wh1   + (size_t)NN * HH;

  gemm3_wmma<<<dim3((NN + 127) / 128), dim3(256), 0, stream>>>(
      feat, W0, B0, W1, B1, W2, B2, selfh, Wh1, Wh2);

  gat_attn_wmma<<<dim3((NN + WV2 - 1) / WV2), dim3(WV2 * 32), 0, stream>>>(
      Wh1, Wh2, selfh, attn, src1, src2, outp);
}
